// Self_Attn_31894427140719
// MI455X (gfx1250) — compile-verified, matrix operand order checked by reading, not run on hardware
//
#include <hip/hip_runtime.h>
#include <hip/hip_fp16.h>

typedef __attribute__((ext_vector_type(16))) _Float16 v16h;
typedef __attribute__((ext_vector_type(8)))  _Float16 v8h;
typedef __attribute__((ext_vector_type(2)))  _Float16 h2;
typedef __attribute__((ext_vector_type(8)))  float    v8f;

#define C_DIM 128
#define CQ    16
#define NSP   4096   // 64*64
#define BATCH 4

// ---------------------------------------------------------------------------
// Projection q,k : written directly in WMMA operand image layout.
//  lane = (c>>3)*16 + (n&15), half j = c&7  (K padded 16->32 with zeros)
// ---------------------------------------------------------------------------
__global__ void proj_qk(const float* __restrict__ x,
                        const float* __restrict__ wq, const float* __restrict__ bq,
                        const float* __restrict__ wk, const float* __restrict__ bk,
                        _Float16* __restrict__ qA, _Float16* __restrict__ kB) {
  int n = blockIdx.x * blockDim.x + threadIdx.x;
  int b = blockIdx.y;
  float accq[CQ], acck[CQ];
#pragma unroll
  for (int i = 0; i < CQ; ++i) { accq[i] = bq[i]; acck[i] = bk[i]; }
  const float* xb = x + ((size_t)b * C_DIM) * NSP + n;
  for (int c = 0; c < C_DIM; ++c) {
    float xv = xb[(size_t)c * NSP];
#pragma unroll
    for (int i = 0; i < CQ; ++i) {
      accq[i] = fmaf(wq[i * C_DIM + c], xv, accq[i]);
      acck[i] = fmaf(wk[i * C_DIM + c], xv, acck[i]);
    }
  }
  int row = n & 15;
  size_t blk = (size_t)b * (NSP / 16) + (n >> 4);
#pragma unroll
  for (int c = 0; c < CQ; ++c) {
    int h = c >> 3, j = c & 7;
    int lane = h * 16 + row;
    qA[(blk * 32 + lane) * 8 + j] = (_Float16)accq[c];
    kB[(blk * 32 + lane) * 8 + j] = (_Float16)acck[c];
  }
}

// ---------------------------------------------------------------------------
// Projection v : stored as WMMA B-operand lane images per (32-key, 16-ch) tile.
// ---------------------------------------------------------------------------
__global__ void proj_v(const float* __restrict__ x, const float* __restrict__ wv,
                       const float* __restrict__ bv, _Float16* __restrict__ vB) {
  int n  = blockIdx.x * blockDim.x + threadIdx.x;
  int cg = blockIdx.y;
  int b  = blockIdx.z;
  float acc[32];
#pragma unroll
  for (int i = 0; i < 32; ++i) acc[i] = bv[cg * 32 + i];
  const float* xb = x + ((size_t)b * C_DIM) * NSP + n;
  for (int c = 0; c < C_DIM; ++c) {
    float xv = xb[(size_t)c * NSP];
#pragma unroll
    for (int i = 0; i < 32; ++i)
      acc[i] = fmaf(wv[(cg * 32 + i) * C_DIM + c], xv, acc[i]);
  }
  int kk = n & 31, kb = n >> 5;
  int h  = (kk >> 3) & 1;
  int hp = (kk & 7) | ((kk >> 4) << 3);
#pragma unroll
  for (int i = 0; i < 32; ++i) {
    int ch = cg * 32 + i, cb = ch >> 4, cc = ch & 15;
    int lane = h * 16 + cc;
    vB[((((size_t)b * (NSP / 32) + kb) * 8 + cb) * 32 + lane) * 16 + hp] = (_Float16)acc[i];
  }
}

// ---------------------------------------------------------------------------
// Flash-attention core, 8 waves/block (each owns 16 query rows).
// Per 64-key block: async-DMA V tile to LDS, 4 energy WMMAs, online softmax
// (packed f16x2 max tree; row-sum comes free from a ones-column WMMA),
// LDS C->A transpose of P, 18 accumulate WMMAs (8 ch blocks x 2 + ones x 2).
// ---------------------------------------------------------------------------
__global__ __launch_bounds__(256) void attn(const _Float16* __restrict__ qA,
                                            const _Float16* __restrict__ kB,
                                            const _Float16* __restrict__ vB,
                                            const float* __restrict__ x,
                                            const float* __restrict__ gamma,
                                            float* __restrict__ out) {
  __shared__ _Float16 lds_p[8][16][64];        // per-wave P staging, 16 KB
  __shared__ _Float16 lds_v[2 * 8 * 32 * 16];  // shared V tile (64 keys), 16 KB

  int b    = blockIdx.y;
  int wid  = threadIdx.x >> 5;
  int lane = threadIdx.x & 31;
  int half = lane >> 4;
  int col  = lane & 15;
  int q16  = blockIdx.x * 8 + wid;             // batch-local 16-row block

  // Query A operand (K = Cq zero-padded to 32)
  v16h aq = {};
  {
    const v8h* src = (const v8h*)(qA + (((size_t)b * (NSP / 16) + q16) * 32 + lane) * 8);
    v8h t = *src;
#pragma unroll
    for (int j = 0; j < 8; ++j) aq[j] = t[j];
  }
  v16h ones;
#pragma unroll
  for (int j = 0; j < 16; ++j) ones[j] = (_Float16)1.0f;

  v8f acc[9];                                  // [0..7]=channels, [8]=rowsum
#pragma unroll
  for (int cb = 0; cb < 9; ++cb) acc[cb] = {};
  float M[8];
#pragma unroll
  for (int r = 0; r < 8; ++r) M[r] = -__builtin_inff();

  unsigned lds_v_base = (unsigned)(size_t)&lds_v[0];

  for (int kb = 0; kb < NSP / 64; ++kb) {
    __syncthreads();   // everyone done reading previous V tile

    // Async DMA: stage this block's V tile (16 KB) into LDS; overlaps the
    // energy WMMAs + softmax below. Tracked by ASYNCcnt.
    {
      unsigned long long gsrc =
          (unsigned long long)(vB + (((size_t)b * (NSP / 32) + kb * 2) * 8) * 32 * 16);
#pragma unroll
      for (int i = 0; i < 4; ++i) {
        unsigned off = (unsigned)(i * 4096 + threadIdx.x * 16);
        asm volatile("global_load_async_to_lds_b128 %0, %1, %2"
                     :: "v"(lds_v_base + off), "v"(off), "s"(gsrc)
                     : "memory");
      }
    }

    // Energy: 4 x 16-key tiles
    v8f E[4];
#pragma unroll
    for (int t = 0; t < 4; ++t) {
      v16h bk16 = {};
      const v8h* s = (const v8h*)(kB + (((size_t)b * (NSP / 16) + kb * 4 + t) * 32 + lane) * 8);
      v8h tt = *s;
#pragma unroll
      for (int j = 0; j < 8; ++j) bk16[j] = tt[j];
      v8f z = {};
      E[t] = __builtin_amdgcn_wmma_f32_16x16x32_f16(false, aq, false, bk16, (short)0, z, false, false);
    }

    // Row max: local 4-way max, then f16x2-packed xor tree across 16 lanes.
    float mx[8];
#pragma unroll
    for (int r = 0; r < 8; ++r)
      mx[r] = fmaxf(fmaxf(E[0][r], E[1][r]), fmaxf(E[2][r], E[3][r]));
#pragma unroll
    for (int jj = 0; jj < 4; ++jj) {
      h2 pk;
      pk[0] = (_Float16)mx[2 * jj];
      pk[1] = (_Float16)mx[2 * jj + 1];
      int pi = __builtin_bit_cast(int, pk);
#pragma unroll
      for (int d = 1; d < 16; d <<= 1) {
        int o = __shfl_xor(pi, d, 32);
        h2 m2 = __builtin_elementwise_max(__builtin_bit_cast(h2, pi),
                                          __builtin_bit_cast(h2, o));
        pi = __builtin_bit_cast(int, m2);
      }
      h2 m2 = __builtin_bit_cast(h2, pi);
      mx[2 * jj]     = (float)m2[0];
      mx[2 * jj + 1] = (float)m2[1];
    }

    // Online-softmax rescale of all 9 accumulators (rowsum included).
    float sc[8];
#pragma unroll
    for (int r = 0; r < 8; ++r) {
      float mn = fmaxf(M[r], mx[r]);
      sc[r] = __expf(M[r] - mn);
      M[r] = mn;
    }
#pragma unroll
    for (int cb = 0; cb < 9; ++cb) {
#pragma unroll
      for (int r = 0; r < 8; ++r) acc[cb][r] *= sc[r];
    }

    // P = exp(E - M) -> LDS (C-layout write)
#pragma unroll
    for (int r = 0; r < 8; ++r) {
      int m = half * 8 + r;
#pragma unroll
      for (int t = 0; t < 4; ++t) {
        float pv = __expf(E[t][r] - M[r]);
        lds_p[wid][m][t * 16 + col] = (_Float16)pv;
      }
    }
    __builtin_amdgcn_wave_barrier();
    asm volatile("s_wait_dscnt 0x0" ::: "memory");

    // V tile must be resident before the accumulate WMMAs.
    asm volatile("s_wait_asynccnt 0x0" ::: "memory");
    __syncthreads();

    // out_tile[q,c] += P x V^T ; rowsum += P x 1
#pragma unroll
    for (int kk = 0; kk < 2; ++kk) {
      v16h p;
      {
        const v8h* lo = (const v8h*)&lds_p[wid][col][kk * 32 + half * 8];
        const v8h* hi = (const v8h*)&lds_p[wid][col][kk * 32 + 16 + half * 8];
        v8h tl = *lo, th = *hi;
#pragma unroll
        for (int j = 0; j < 8; ++j) { p[j] = tl[j]; p[j + 8] = th[j]; }
      }
#pragma unroll
      for (int cb = 0; cb < 8; ++cb) {
        const v8h* vv = (const v8h*)&lds_v[((kk * 8 + cb) * 32 + lane) * 16];
        v8h t0 = vv[0], t1 = vv[1];
        v16h bv16;
#pragma unroll
        for (int j = 0; j < 8; ++j) { bv16[j] = t0[j]; bv16[j + 8] = t1[j]; }
        acc[cb] = __builtin_amdgcn_wmma_f32_16x16x32_f16(false, p, false, bv16, (short)0, acc[cb], false, false);
      }
      acc[8] = __builtin_amdgcn_wmma_f32_16x16x32_f16(false, p, false, ones, (short)0, acc[8], false, false);
    }
  }

  // Epilogue: normalize by rowsum (acc[8]), gamma, residual.
  float g = gamma[0];
  int nbase = q16 * 16;
#pragma unroll
  for (int r = 0; r < 8; ++r) {
    float inv = 1.0f / acc[8][r];
    int m = half * 8 + r;
    int n = nbase + m;
#pragma unroll
    for (int cb = 0; cb < 8; ++cb) {
      int c = cb * 16 + col;
      size_t idx = ((size_t)b * C_DIM + c) * NSP + n;
      out[idx] = g * (acc[cb][r] * inv) + x[idx];
    }
  }
}

// ---------------------------------------------------------------------------
extern "C" void kernel_launch(void* const* d_in, const int* in_sizes, int n_in,
                              void* d_out, int out_size, void* d_ws, size_t ws_size,
                              hipStream_t stream) {
  const float* x     = (const float*)d_in[0];
  const float* wq    = (const float*)d_in[1];
  const float* bq    = (const float*)d_in[2];
  const float* wk    = (const float*)d_in[3];
  const float* bk    = (const float*)d_in[4];
  const float* wv    = (const float*)d_in[5];
  const float* bv    = (const float*)d_in[6];
  const float* gamma = (const float*)d_in[7];
  float* out = (float*)d_out;

  _Float16* qA = (_Float16*)d_ws;
  _Float16* kB = qA + (size_t)BATCH * (NSP / 16) * 32 * 8;
  _Float16* vB = kB + (size_t)BATCH * (NSP / 16) * 32 * 8;

  proj_qk<<<dim3(NSP / 256, BATCH), 256, 0, stream>>>(x, wq, bq, wk, bk, qA, kB);
  proj_v <<<dim3(NSP / 256, 4, BATCH), 256, 0, stream>>>(x, wv, bv, vB);
  attn   <<<dim3(NSP / 128, BATCH), 256, 0, stream>>>(qA, kB, vB, x, gamma, out);
}
